// MultiHeadEinsumAttention_14104672600820
// MI455X (gfx1250) — compile-verified, matrix operand order checked by reading, not run on hardware
//
#include <hip/hip_runtime.h>
#include <hip/hip_bf16.h>

// ---------------------------------------------------------------------------
// MHA forward for MI455X (gfx1250, wave32, WMMA).
//   cast/transpose -> QKV GEMM (32x64/wave, wmma) -> flash attention
//   (32 queries/wave, wmma) -> projection GEMM (32x64/wave, wmma, +bias)
// All matmuls: v_wmma_f32_16x16x32_f16 (f16 in, f32 accumulate).
// Fragment loads use ONE lane-base pointer per matrix + constant offsets so
// they fold into global_load_b128 immediate offsets (no per-row pointers).
// ---------------------------------------------------------------------------

#define N_EMBD    1024
#define NUM_HEADS 16
#define HEAD_SIZE 64
#define BATCH     2
#define SEQ       2048
#define BT        (BATCH * SEQ)   // 4096 rows of x
#define QKV_N     (3 * N_EMBD)    // 3072

typedef __attribute__((ext_vector_type(16))) _Float16 v16h;
typedef __attribute__((ext_vector_type(8)))  _Float16 v8h;
typedef __attribute__((ext_vector_type(8)))  float    v8f;

// Load a 16x32 f16 fragment from a pointer that already encodes:
//   matrix base + lane_row*ld + ((lane&16)?8:0) + k0 (+ constant tile offset).
// Fragment K-mapping (per ISA 16-bit A layout):
//   lanes 0-15 : halves 0-7 -> K=+0..7,  halves 8-15 -> K=+16..23
//   lanes 16-31: halves 0-7 -> K=+8..15, halves 8-15 -> K=+24..31
static __device__ inline v16h load_frag_at(const _Float16* __restrict__ p) {
  v8h lo = *(const v8h*)(p);
  v8h hi = *(const v8h*)(p + 16);
  v16h r;
#pragma unroll
  for (int i = 0; i < 8; ++i) { r[i] = lo[i]; r[i + 8] = hi[i]; }
  return r;
}

static __device__ inline v8f wmma_f16(v16h a, v16h b, v8f c) {
  return __builtin_amdgcn_wmma_f32_16x16x32_f16(false, a, false, b,
                                                (short)0, c, false, false);
}

// ---------------------------------------------------------------------------
// Conversion kernels
// ---------------------------------------------------------------------------
__global__ void cast_f32_to_f16(const float* __restrict__ in,
                                _Float16* __restrict__ out, int n) {
  int i = blockIdx.x * blockDim.x + threadIdx.x;
  if (i < n) out[i] = (_Float16)in[i];
}

// in [K,N] f32 (row-major) -> out [N,K] f16 (row-major transpose)
__global__ void transpose_f32_to_f16(const float* __restrict__ in,
                                     _Float16* __restrict__ out, int K, int N) {
  int i = blockIdx.x * blockDim.x + threadIdx.x;
  if (i < K * N) {
    int k = i / N, n = i % N;
    out[(long)n * K + k] = (_Float16)in[i];
  }
}

// ---------------------------------------------------------------------------
// QKV projection: qkv[4096,3072] = xh @ w_qkv, scattered into
//   Q [B,H,T,D] (x 1/sqrt(D)), K [B,H,T,D], V^T [B,H,D,T]  (all f16).
// Each wave: 32x64 macro-tile (2 A frags + 4 B frags -> 8 wmma per k-step).
// ---------------------------------------------------------------------------
__global__ void qkv_gemm(const _Float16* __restrict__ xh,
                         const _Float16* __restrict__ wT,   // [3072,1024]
                         _Float16* __restrict__ Qh,
                         _Float16* __restrict__ Kh,
                         _Float16* __restrict__ VTh) {
  const int lane = threadIdx.x & 31;
  const int wave = threadIdx.x >> 5;
  const int tid  = blockIdx.x * 8 + wave;
  const int NT   = QKV_N / 64;                 // 48 macro-tiles along N
  const int mt   = tid / NT, nt = tid % NT;
  const int m0   = mt * 32, n0 = nt * 64;
  const int l15  = lane & 15;
  const int koff = (lane & 16) ? 8 : 0;

  // One lane-base pointer per matrix; all fragment rows are constant offsets.
  const _Float16* abase = xh + (long)(m0 + l15) * N_EMBD + koff;
  const _Float16* bbase = wT + (long)(n0 + l15) * N_EMBD + koff;

  v8f acc[2][4] = {};
#pragma unroll 2
  for (int k0 = 0; k0 < N_EMBD; k0 += 32) {
    const _Float16* ap = abase + k0;
    const _Float16* bp = bbase + k0;
    if (k0 + 32 < N_EMBD) {                    // pull next k-block toward L1
      __builtin_prefetch(bp + 32, 0, 1);
      __builtin_prefetch(ap + 32, 0, 1);
    }
    v16h a[2], b[4];
#pragma unroll
    for (int i = 0; i < 2; ++i)
      a[i] = load_frag_at(ap + i * 16 * N_EMBD);   // +0, +32KB
#pragma unroll
    for (int j = 0; j < 4; ++j)
      b[j] = load_frag_at(bp + j * 16 * N_EMBD);   // +0..+96KB
#pragma unroll
    for (int i = 0; i < 2; ++i)
#pragma unroll
      for (int j = 0; j < 4; ++j)
        acc[i][j] = wmma_f16(a[i], b[j], acc[i][j]);
  }

  // Column decomposition: c = h*192 + d*3 + three ('three' innermost).
  const int hsel = koff;                       // 8 iff upper half-wave
#pragma unroll
  for (int j = 0; j < 4; ++j) {
    const int c     = n0 + j * 16 + l15;       // per-lane output column
    const int three = c % 3;
    const int d     = (c / 3) & (HEAD_SIZE - 1);
    const int h     = c / (HEAD_SIZE * 3);
#pragma unroll
    for (int i = 0; i < 2; ++i) {
      const int mb = m0 + i * 16 + hsel;
#pragma unroll
      for (int r = 0; r < 8; ++r) {
        const int m  = mb + r;
        const int bb = m >> 11, t = m & (SEQ - 1);   // m / SEQ, m % SEQ
        const long bh = (long)(bb * NUM_HEADS + h);
        const float v = acc[i][j][r];
        if (three == 0)
          Qh[(bh * SEQ + t) * HEAD_SIZE + d] = (_Float16)(v * 0.125f);
        else if (three == 1)
          Kh[(bh * SEQ + t) * HEAD_SIZE + d] = (_Float16)v;
        else
          VTh[(bh * HEAD_SIZE + d) * SEQ + t] = (_Float16)v;
      }
    }
  }
}

// ---------------------------------------------------------------------------
// Flash attention: one wave per 32-query tile of one (b,h).  K/V fragments
// are shared by both 16-row tiles: 8 loads feed 16 wmma per 32-key block.
// ---------------------------------------------------------------------------
__global__ void flash_attn(const _Float16* __restrict__ Qh,
                           const _Float16* __restrict__ Kh,
                           const _Float16* __restrict__ VTh,
                           _Float16* __restrict__ AO) {   // [B*T, C] f16
  __shared__ __align__(16) _Float16 psh[4][2][16][32];    // [wave][rowtile]

  const int lane = threadIdx.x & 31;
  const int wave = threadIdx.x >> 5;
  const int wtid = blockIdx.x * 4 + wave;
  const int qt   = wtid & (SEQ / 32 - 1);      // 32-query tile 0..63
  const int bh   = wtid >> 6;                  // 0..31
  const int i0   = qt * 32;
  const int l15  = lane & 15;
  const int koff = (lane & 16) ? 8 : 0;
  const int hsel = koff;

  // Lane-base pointers (fold in lane row and K sub-offset once).
  const _Float16* qb = Qh  + (long)bh * SEQ * HEAD_SIZE
                     + (long)(i0 + l15) * HEAD_SIZE + koff;
  const _Float16* kb = Kh  + (long)bh * SEQ * HEAD_SIZE
                     + (long)l15 * HEAD_SIZE + koff;
  const _Float16* vb = VTh + (long)bh * HEAD_SIZE * SEQ
                     + (long)l15 * SEQ + koff;

  v16h aq[2][2];                               // [rowtile][k-half]
#pragma unroll
  for (int i = 0; i < 2; ++i)
#pragma unroll
    for (int p = 0; p < 2; ++p)
      aq[i][p] = load_frag_at(qb + i * 16 * HEAD_SIZE + p * 32);

  v8f o[2][4] = {};
  float mst[2][8], lst[2][8];
#pragma unroll
  for (int i = 0; i < 2; ++i)
#pragma unroll
    for (int r = 0; r < 8; ++r) { mst[i][r] = -1e30f; lst[i][r] = 0.f; }

  for (int j = 0; j <= qt; ++j) {              // causal: qt+1 key blocks
    const int j0 = j * 32;
    const _Float16* kp = kb + (long)j0 * HEAD_SIZE;  // advances 4KB per block
    const _Float16* vp = vb + j0;                    // advances 64B per block

    v16h bk[2][2];                             // [key-subtile][k-half]
#pragma unroll
    for (int u = 0; u < 2; ++u)
#pragma unroll
      for (int p = 0; p < 2; ++p)
        bk[u][p] = load_frag_at(kp + u * 16 * HEAD_SIZE + p * 32);

    v8f s[2][2] = {};                          // [rowtile][key-subtile]
#pragma unroll
    for (int i = 0; i < 2; ++i)
#pragma unroll
      for (int u = 0; u < 2; ++u) {
        s[i][u] = wmma_f16(aq[i][0], bk[u][0], s[i][u]);
        s[i][u] = wmma_f16(aq[i][1], bk[u][1], s[i][u]);
      }

    const int kn0 = j0 + l15;                  // key index of s[.][0] columns
#pragma unroll
    for (int i = 0; i < 2; ++i) {
#pragma unroll
      for (int r = 0; r < 8; ++r) {
        const int qm = i0 + i * 16 + hsel + r;
        float v0 = (kn0      <= qm) ? s[i][0][r] : -1e30f;  // causal mask
        float v1 = (kn0 + 16 <= qm) ? s[i][1][r] : -1e30f;
        float rm = fmaxf(v0, v1);
        rm = fmaxf(rm, __shfl_xor(rm, 1, 32));
        rm = fmaxf(rm, __shfl_xor(rm, 2, 32));
        rm = fmaxf(rm, __shfl_xor(rm, 4, 32));
        rm = fmaxf(rm, __shfl_xor(rm, 8, 32));
        const float mnew  = fmaxf(mst[i][r], rm);
        const float alpha = __expf(mst[i][r] - mnew);
        const float p0 = __expf(v0 - mnew);
        const float p1 = __expf(v1 - mnew);
        float rs = p0 + p1;
        rs += __shfl_xor(rs, 1, 32);
        rs += __shfl_xor(rs, 2, 32);
        rs += __shfl_xor(rs, 4, 32);
        rs += __shfl_xor(rs, 8, 32);
        lst[i][r] = lst[i][r] * alpha + rs;
        mst[i][r] = mnew;
#pragma unroll
        for (int dch = 0; dch < 4; ++dch) o[i][dch][r] *= alpha;
        psh[wave][i][hsel + r][l15]      = (_Float16)p0;    // LDS transpose
        psh[wave][i][hsel + r][l15 + 16] = (_Float16)p1;
      }
    }

    // Reload P tiles as A-fragments (same wave; LDS is in-order per wave).
    v16h pa[2];
    {
      const _Float16* pb = &psh[wave][0][l15][koff];
#pragma unroll
      for (int i = 0; i < 2; ++i) {
        const _Float16* pp = pb + i * 16 * 32;      // rowtile stride: 1KB
        v8h lo = *(const v8h*)(pp);
        v8h hi = *(const v8h*)(pp + 16);
#pragma unroll
        for (int q = 0; q < 8; ++q) { pa[i][q] = lo[q]; pa[i][q + 8] = hi[q]; }
      }
    }

    v16h bv[4];                                // V fragments from V^T
#pragma unroll
    for (int dch = 0; dch < 4; ++dch)
      bv[dch] = load_frag_at(vp + (long)dch * 16 * SEQ);   // +0..+192KB
#pragma unroll
    for (int i = 0; i < 2; ++i)
#pragma unroll
      for (int dch = 0; dch < 4; ++dch)
        o[i][dch] = wmma_f16(pa[i], bv[dch], o[i][dch]);
  }

  // Epilogue: O /= l, store to AO[b*T+t][h*64+d]  (b h t d -> b t (h d)).
  const int h  = bh & (NUM_HEADS - 1);
  const int bb = bh >> 4;
#pragma unroll
  for (int i = 0; i < 2; ++i)
#pragma unroll
    for (int r = 0; r < 8; ++r) {
      const int   m    = i0 + i * 16 + hsel + r;
      const float inv  = 1.f / lst[i][r];
      const long  base = ((long)(bb * SEQ + m)) * N_EMBD + h * HEAD_SIZE + l15;
      AO[base +  0] = (_Float16)(o[i][0][r] * inv);
      AO[base + 16] = (_Float16)(o[i][1][r] * inv);
      AO[base + 32] = (_Float16)(o[i][2][r] * inv);
      AO[base + 48] = (_Float16)(o[i][3][r] * inv);
    }
}

// ---------------------------------------------------------------------------
// Output projection: out[4096,1024] (f32) = AO @ w_proj + b_proj.
// Same 32x64 per-wave macro-tile.
// ---------------------------------------------------------------------------
__global__ void proj_gemm(const _Float16* __restrict__ aoh,
                          const _Float16* __restrict__ wT,   // [1024,1024]
                          const float* __restrict__ bias,
                          float* __restrict__ out) {
  const int lane = threadIdx.x & 31;
  const int wave = threadIdx.x >> 5;
  const int tid  = blockIdx.x * 8 + wave;
  const int NT   = N_EMBD / 64;                // 16 macro-tiles along N
  const int mt   = tid / NT, nt = tid % NT;
  const int m0   = mt * 32, n0 = nt * 64;
  const int l15  = lane & 15;
  const int koff = (lane & 16) ? 8 : 0;

  const _Float16* abase = aoh + (long)(m0 + l15) * N_EMBD + koff;
  const _Float16* bbase = wT  + (long)(n0 + l15) * N_EMBD + koff;

  v8f acc[2][4] = {};
#pragma unroll 2
  for (int k0 = 0; k0 < N_EMBD; k0 += 32) {
    const _Float16* ap = abase + k0;
    const _Float16* bp = bbase + k0;
    if (k0 + 32 < N_EMBD) {
      __builtin_prefetch(bp + 32, 0, 1);
      __builtin_prefetch(ap + 32, 0, 1);
    }
    v16h a[2], b[4];
#pragma unroll
    for (int i = 0; i < 2; ++i)
      a[i] = load_frag_at(ap + i * 16 * N_EMBD);
#pragma unroll
    for (int j = 0; j < 4; ++j)
      b[j] = load_frag_at(bp + j * 16 * N_EMBD);
#pragma unroll
    for (int i = 0; i < 2; ++i)
#pragma unroll
      for (int j = 0; j < 4; ++j)
        acc[i][j] = wmma_f16(a[i], b[j], acc[i][j]);
  }

  const int hsel = koff;
#pragma unroll
  for (int j = 0; j < 4; ++j) {
    const int   n  = n0 + j * 16 + l15;
    const float bn = bias[n];
#pragma unroll
    for (int i = 0; i < 2; ++i) {
      const int mb = m0 + i * 16 + hsel;
#pragma unroll
      for (int r = 0; r < 8; ++r)
        out[(long)(mb + r) * N_EMBD + n] = acc[i][j][r] + bn;
    }
  }
}

// ---------------------------------------------------------------------------
extern "C" void kernel_launch(void* const* d_in, const int* in_sizes, int n_in,
                              void* d_out, int out_size, void* d_ws, size_t ws_size,
                              hipStream_t stream) {
  const float* x      = (const float*)d_in[0];
  const float* w_qkv  = (const float*)d_in[1];
  const float* w_proj = (const float*)d_in[2];
  const float* b_proj = (const float*)d_in[3];
  float* out = (float*)d_out;

  // Workspace carve-up (f16): ~48 MB total, resident in the 192 MB L2.
  char* ws = (char*)d_ws;
  _Float16* xh     = (_Float16*)ws; ws += (size_t)BT * N_EMBD * 2;      // 8 MB
  _Float16* wqkvT  = (_Float16*)ws; ws += (size_t)QKV_N * N_EMBD * 2;   // 6 MB
  _Float16* wprojT = (_Float16*)ws; ws += (size_t)N_EMBD * N_EMBD * 2;  // 2 MB
  _Float16* Qh     = (_Float16*)ws; ws += (size_t)BT * N_EMBD * 2;      // 8 MB
  _Float16* Kh     = (_Float16*)ws; ws += (size_t)BT * N_EMBD * 2;      // 8 MB
  _Float16* VTh    = (_Float16*)ws; ws += (size_t)BT * N_EMBD * 2;      // 8 MB
  _Float16* AO     = (_Float16*)ws; ws += (size_t)BT * N_EMBD * 2;      // 8 MB
  (void)in_sizes; (void)n_in; (void)out_size; (void)ws_size;

  {
    int n = BT * N_EMBD;
    cast_f32_to_f16<<<(n + 255) / 256, 256, 0, stream>>>(x, xh, n);
  }
  transpose_f32_to_f16<<<(N_EMBD * QKV_N + 255) / 256, 256, 0, stream>>>(
      w_qkv, wqkvT, N_EMBD, QKV_N);
  transpose_f32_to_f16<<<(N_EMBD * N_EMBD + 255) / 256, 256, 0, stream>>>(
      w_proj, wprojT, N_EMBD, N_EMBD);

  // (4096/32)*(3072/64) = 6144 wave-tiles / 8 waves per block
  qkv_gemm<<<(BT / 32) * (QKV_N / 64) / 8, 256, 0, stream>>>(xh, wqkvT, Qh, Kh, VTh);

  // B*H*(T/32) = 2048 wave-tiles / 4 waves per block
  flash_attn<<<(BATCH * NUM_HEADS * (SEQ / 32)) / 4, 128, 0, stream>>>(Qh, Kh, VTh, AO);

  // (4096/32)*(1024/64) = 2048 wave-tiles / 8 waves per block
  proj_gemm<<<(BT / 32) * (N_EMBD / 64) / 8, 256, 0, stream>>>(AO, wprojT, b_proj, out);
}
